// TransformerBlock_489626272035
// MI455X (gfx1250) — compile-verified, matrix operand order checked by reading, not run on hardware
//
#include <hip/hip_runtime.h>
#include <math.h>

#define BB    4
#define TT    1024
#define EMB   2048
#define QHN   32
#define KVHN  8
#define HDIM  64
#define FFND  8192
#define MTOK  (BB * TT)      /* 4096 tokens */
#define KVD   (KVHN * HDIM)  /* 512 */

typedef __bf16 bf16;
typedef __attribute__((ext_vector_type(16))) __bf16 v16bf;
typedef __attribute__((ext_vector_type(8)))  __bf16 v8bf;
typedef __attribute__((ext_vector_type(8)))  float  v8f;
typedef __attribute__((ext_vector_type(4)))  unsigned int u32x4;
typedef __attribute__((ext_vector_type(8)))  unsigned int u32x8;

static __device__ __forceinline__ v8f zero8() {
  v8f z;
#pragma unroll
  for (int i = 0; i < 8; ++i) z[i] = 0.0f;
  return z;
}

// Build a 16-element bf16 WMMA fragment from two contiguous 8-element (16B) chunks.
// CDNA5 16-bit A/B layout: elements 0..7 -> K = koff..koff+7,
// elements 8..15 -> K = 16+koff..16+koff+7 (koff = 8*(lane>=16)).
static __device__ __forceinline__ v16bf ldfrag(const bf16* lo, const bf16* hi) {
  union { v16bf v; v8bf h[2]; } u;
  u.h[0] = *(const v8bf*)lo;
  u.h[1] = *(const v8bf*)hi;
  return u.v;
}

// ---------------------------------------------------------------- TDM -------
// Issue one Tensor-Data-Mover DMA: a [rows x 32] bf16 tile whose source rows are
// strideElems apart, landing in LDS with 16B padding every 64B (-> 80B LDS rows,
// matching the [..][40] bf16 shared layout). Tracked by TENSORcnt.
static __device__ __forceinline__ void tdm_load_tile(const bf16* gaddr,
                                                     unsigned lds_off,
                                                     unsigned strideElems,
                                                     unsigned rows) {
  unsigned long long ga = (unsigned long long)(uintptr_t)gaddr;
  u32x4 g0;
  g0[0] = 1u;                                        // count=1, user descriptor
  g0[1] = lds_off;                                   // lds_addr (bytes)
  g0[2] = (unsigned)(ga & 0xFFFFFFFFu);              // global_addr[31:0]
  g0[3] = (unsigned)((ga >> 32) & 0x01FFFFFFu)       // global_addr[56:32]
          | (2u << 30);                              // type = 2 ("image")
  u32x8 g1;
  g1[0] = (1u << 16)        // data_size = 2 bytes
        | (1u << 20)        // pad_enable
        | (3u << 22)        // pad_interval: 16 DWORDs (64B) between pads
        | (3u << 25);       // pad_amount: 4 DWORDs (16B)
  g1[1] = 32u << 16;                      // tensor_dim0 = 32 (tile-sized, no OOB)
  g1[2] = (rows & 0xFFFFu) << 16;         // tensor_dim1 = rows
  g1[3] = 32u << 16;                      // tile_dim0 = 32
  g1[4] = rows & 0xFFFFu;                 // tile_dim1 = rows, tile_dim2 = 0
  g1[5] = strideElems;                    // tensor_dim0_stride (elements)
  g1[6] = 0u;
  g1[7] = 0u;
  asm volatile("tensor_load_to_lds %0, %1" :: "s"(g0), "s"(g1) : "memory");
}

static __device__ __forceinline__ unsigned lds_off_of(const void* p) {
  return (unsigned)(uintptr_t)p;   // generic shared ptr: low 32 bits = LDS offset
}

// ---------------------------------------------------------------- converts ---
__global__ void cvt_f32_bf16(const float* __restrict__ in, bf16* __restrict__ out, long n) {
  long i = (long)blockIdx.x * blockDim.x + threadIdx.x;
  long stride = (long)gridDim.x * blockDim.x;
  for (; i < n; i += stride) out[i] = (bf16)in[i];
}

// W is [K][N] fp32 row-major; Wt is [N][K] bf16 row-major (K contiguous).
__global__ void cvt_transpose(const float* __restrict__ W, bf16* __restrict__ Wt,
                              int K, int N) {
  long total = (long)K * N;
  long i = (long)blockIdx.x * blockDim.x + threadIdx.x;
  long stride = (long)gridDim.x * blockDim.x;
  for (; i < total; i += stride) {
    long nIdx = i / K;
    long k    = i - nIdx * K;
    Wt[i] = (bf16)W[k * (long)N + nIdx];
  }
}

// ------------------------------------------------------------------- GEMM ----
// C[M][N] = A[M][K](bf16) * Bt[N][K](bf16)^T, fp32 accumulate.
// Block tile 128x256, 8 waves, each wave 64x64 (16 WMMAs per K-step).
// A/B tiles staged by TDM DMA with double buffering (TENSORcnt).
// mode 0: Cf = acc + bias
// mode 1: Cb = bf16(acc + bias)
// mode 2: Cb = bf16(gelu_exact(acc + bias))
// mode 3: Cf = acc + bias + resid
__global__ __launch_bounds__(256)
void gemm_bf16(const bf16* __restrict__ A, const bf16* __restrict__ Bt,
               const float* __restrict__ bias, const float* __restrict__ resid,
               float* __restrict__ Cf, bf16* __restrict__ Cb,
               int Nd, int Kd, int mode) {
  __shared__ bf16 As[2][128][40];   // 32 cols used, 80B rows (TDM-padded)
  __shared__ bf16 Bs[2][256][40];

  const int tid    = threadIdx.x;
  const int lane   = tid & 31;
  const int wave   = tid >> 5;
  const int half   = lane >> 4;
  const int lanelo = lane & 15;
  const int koff   = half * 8;
  const int wm     = wave >> 2;         // 0..1 -> 64 rows each
  const int wn     = wave & 3;          // 0..3 -> 64 cols each
  const int m0     = blockIdx.y * 128;
  const int n0     = blockIdx.x * 256;

  v8f acc[4][4];
#pragma unroll
  for (int mi = 0; mi < 4; ++mi)
#pragma unroll
    for (int ni = 0; ni < 4; ++ni) acc[mi][ni] = zero8();

  // prologue DMA: tile kt=0 into buffer 0
  if (wave == 0) {
    tdm_load_tile(A  + (size_t)m0 * Kd, lds_off_of(&As[0][0][0]), (unsigned)Kd, 128u);
    tdm_load_tile(Bt + (size_t)n0 * Kd, lds_off_of(&Bs[0][0][0]), (unsigned)Kd, 256u);
  }

  int pb = 0;
  for (int kt = 0; kt < Kd; kt += 32) {
    if (wave == 0) __builtin_amdgcn_s_wait_tensorcnt(0);
    __syncthreads();                       // buffer pb ready; pb^1 free

    if (wave == 0 && kt + 32 < Kd) {       // prefetch next tile via TDM
      tdm_load_tile(A  + (size_t)m0 * Kd + kt + 32,
                    lds_off_of(&As[pb ^ 1][0][0]), (unsigned)Kd, 128u);
      tdm_load_tile(Bt + (size_t)n0 * Kd + kt + 32,
                    lds_off_of(&Bs[pb ^ 1][0][0]), (unsigned)Kd, 256u);
    }

    v16bf afr[4];
#pragma unroll
    for (int mi = 0; mi < 4; ++mi) {
      const bf16* p = &As[pb][wm * 64 + mi * 16 + lanelo][0];
      afr[mi] = ldfrag(p + koff, p + 16 + koff);
    }
#pragma unroll
    for (int ni = 0; ni < 4; ++ni) {
      const bf16* p = &Bs[pb][wn * 64 + ni * 16 + lanelo][0];
      v16bf bfr = ldfrag(p + koff, p + 16 + koff);
#pragma unroll
      for (int mi = 0; mi < 4; ++mi)
        acc[mi][ni] = __builtin_amdgcn_wmma_f32_16x16x32_bf16(
            false, afr[mi], false, bfr, (short)0, acc[mi][ni], false, false);
    }
    __syncthreads();
    pb ^= 1;
  }

#pragma unroll
  for (int mi = 0; mi < 4; ++mi) {
#pragma unroll
    for (int ni = 0; ni < 4; ++ni) {
      const int col = n0 + wn * 64 + ni * 16 + lanelo;
      const float bv = bias[col];
#pragma unroll
      for (int r = 0; r < 8; ++r) {
        const int row = m0 + wm * 64 + mi * 16 + r + 8 * half;
        float v = acc[mi][ni][r] + bv;
        const size_t idx = (size_t)row * Nd + col;
        if (mode == 0)      Cf[idx] = v;
        else if (mode == 1) Cb[idx] = (bf16)v;
        else if (mode == 2) Cb[idx] = (bf16)(0.5f * v * (1.0f + erff(v * 0.70710678118f)));
        else                Cf[idx] = v + resid[idx];
      }
    }
  }
}

// ------------------------------------------------------------------ rotary ---
__global__ void rotary_bf16(const float* __restrict__ src, bf16* __restrict__ dst,
                            int heads) {
  const int hh = heads * (HDIM / 2);
  long total = (long)MTOK * hh;
  long i = (long)blockIdx.x * blockDim.x + threadIdx.x;
  long stride = (long)gridDim.x * blockDim.x;
  for (; i < total; i += stride) {
    long row = i / hh;
    int rem  = (int)(i - row * hh);
    int head = rem / (HDIM / 2);
    int p    = rem % (HDIM / 2);
    int t    = (int)(row % TT);
    float inv = __powf(10000.0f, -((float)(2 * p)) / (float)HDIM);
    float ang = (float)t * inv;
    float c = __cosf(ang), s = __sinf(ang);
    size_t base = (size_t)row * heads * HDIM + (size_t)head * HDIM + 2 * p;
    float x1 = src[base], x2 = src[base + 1];
    dst[base]     = (bf16)(x1 * c - x2 * s);
    dst[base + 1] = (bf16)(x1 * s + x2 * c);
  }
}

// -------------------------------------------------------------- attention ---
// One block = one (b,h), 128 q rows; 8 waves x 16 q rows. 32-key chunks.
__global__ __launch_bounds__(256)
void flash_attn(const bf16* __restrict__ qb, const bf16* __restrict__ kb,
                const bf16* __restrict__ vb, bf16* __restrict__ ctxb) {
  __shared__ bf16 Kc[32][64];        // [key][d]   (Bt layout for Q@K^T)
  __shared__ bf16 Vt[64][32];        // [d][key]   (Bt layout for P@V)
  __shared__ bf16 Pws[8][16][32];    // per-wave P transpose scratch

  const int bh   = blockIdx.y;
  const int b    = bh / QHN;
  const int h    = bh % QHN;
  const int kvh  = h / (QHN / KVHN);
  const int qbase = blockIdx.x * 128;
  const int tid    = threadIdx.x;
  const int lane   = tid & 31;
  const int wave   = tid >> 5;
  const int half   = lane >> 4;
  const int lanelo = lane & 15;
  const int koff   = half * 8;

  const bf16* qrow = qb + ((size_t)(b * TT + qbase + wave * 16 + lanelo)) * EMB + h * HDIM;
  const v16bf qa0 = ldfrag(qrow + koff,      qrow + 16 + koff);
  const v16bf qa1 = ldfrag(qrow + 32 + koff, qrow + 48 + koff);

  v8f o[4];
  float mrow[8], lrow[8];
#pragma unroll
  for (int ni = 0; ni < 4; ++ni) o[ni] = zero8();
#pragma unroll
  for (int r = 0; r < 8; ++r) { mrow[r] = -INFINITY; lrow[r] = 0.0f; }

  const int sj = tid >> 3;          // 0..31 key within chunk
  const int sd = (tid & 7) * 8;     // d block

  for (int kc = 0; kc < TT; kc += 32) {
    __syncthreads();
    {
      const size_t krow = ((size_t)(b * TT + kc + sj)) * KVD + kvh * HDIM + sd;
      *(v8bf*)&Kc[sj][sd] = *(const v8bf*)(kb + krow);
      v8bf vv = *(const v8bf*)(vb + krow);
#pragma unroll
      for (int e = 0; e < 8; ++e) Vt[sd + e][sj] = vv[e];
    }
    __syncthreads();

    v8f s0 = zero8(), s1 = zero8();
    {
      const bf16* p0 = &Kc[lanelo][0];
      v16bf k00 = ldfrag(p0 + koff,      p0 + 16 + koff);
      v16bf k01 = ldfrag(p0 + 32 + koff, p0 + 48 + koff);
      s0 = __builtin_amdgcn_wmma_f32_16x16x32_bf16(false, qa0, false, k00, (short)0, s0, false, false);
      s0 = __builtin_amdgcn_wmma_f32_16x16x32_bf16(false, qa1, false, k01, (short)0, s0, false, false);
      const bf16* p1 = &Kc[16 + lanelo][0];
      v16bf k10 = ldfrag(p1 + koff,      p1 + 16 + koff);
      v16bf k11 = ldfrag(p1 + 32 + koff, p1 + 48 + koff);
      s1 = __builtin_amdgcn_wmma_f32_16x16x32_bf16(false, qa0, false, k10, (short)0, s1, false, false);
      s1 = __builtin_amdgcn_wmma_f32_16x16x32_bf16(false, qa1, false, k11, (short)0, s1, false, false);
    }

#pragma unroll
    for (int r = 0; r < 8; ++r) {
      float a0 = s0[r] * 0.125f;   // 1/sqrt(64)
      float a1 = s1[r] * 0.125f;
      float rm = fmaxf(a0, a1);
#pragma unroll
      for (int off = 1; off < 16; off <<= 1)
        rm = fmaxf(rm, __shfl_xor(rm, off, 16));
      float mn = fmaxf(mrow[r], rm);
      float corr = __expf(mrow[r] - mn);
      float p0v = __expf(a0 - mn);
      float p1v = __expf(a1 - mn);
      float rs = p0v + p1v;
#pragma unroll
      for (int off = 1; off < 16; off <<= 1)
        rs += __shfl_xor(rs, off, 16);
      lrow[r] = lrow[r] * corr + rs;
      mrow[r] = mn;
#pragma unroll
      for (int ni = 0; ni < 4; ++ni) o[ni][r] *= corr;
      Pws[wave][r + 8 * half][lanelo]      = (bf16)p0v;
      Pws[wave][r + 8 * half][16 + lanelo] = (bf16)p1v;
    }
    __syncthreads();

    const bf16* pp = &Pws[wave][lanelo][0];
    v16bf pa = ldfrag(pp + koff, pp + 16 + koff);
#pragma unroll
    for (int ni = 0; ni < 4; ++ni) {
      const bf16* vp = &Vt[ni * 16 + lanelo][0];
      v16bf vf = ldfrag(vp + koff, vp + 16 + koff);
      o[ni] = __builtin_amdgcn_wmma_f32_16x16x32_bf16(false, pa, false, vf, (short)0, o[ni], false, false);
    }
  }

#pragma unroll
  for (int ni = 0; ni < 4; ++ni) {
#pragma unroll
    for (int r = 0; r < 8; ++r) {
      const int trow = qbase + wave * 16 + r + 8 * half;
      float v = o[ni][r] / lrow[r];
      ctxb[((size_t)(b * TT + trow)) * EMB + h * HDIM + ni * 16 + lanelo] = (bf16)v;
    }
  }
}

// ---------------------------------------------------------------- rmsnorm ---
__global__ __launch_bounds__(256)
void rmsnorm_k(const float* __restrict__ x, const float* __restrict__ g,
               float* __restrict__ of, bf16* __restrict__ ob, int cols) {
  __shared__ float red[8];
  const int row = blockIdx.x;
  const float* xr = x + (size_t)row * cols;
  float ss = 0.0f;
  for (int c = threadIdx.x; c < cols; c += 256) { float v = xr[c]; ss += v * v; }
#pragma unroll
  for (int off = 1; off < 32; off <<= 1) ss += __shfl_xor(ss, off, 32);
  if ((threadIdx.x & 31) == 0) red[threadIdx.x >> 5] = ss;
  __syncthreads();
  float tot = 0.0f;
#pragma unroll
  for (int w = 0; w < 8; ++w) tot += red[w];
  const float norm  = sqrtf(tot);
  const float scale = 1.0f / (norm / sqrtf((float)cols) + 1e-6f);
  for (int c = threadIdx.x; c < cols; c += 256) {
    float v = g[c] * xr[c] * scale;
    of[(size_t)row * cols + c] = v;
    if (ob) ob[(size_t)row * cols + c] = (bf16)v;
  }
}

// ----------------------------------------------------------------- launch ---
extern "C" void kernel_launch(void* const* d_in, const int* in_sizes, int n_in,
                              void* d_out, int out_size, void* d_ws, size_t ws_size,
                              hipStream_t stream) {
  (void)in_sizes; (void)n_in; (void)out_size; (void)ws_size;
  const float* x  = (const float*)d_in[0];
  const float* Wq = (const float*)d_in[1];
  const float* bq = (const float*)d_in[2];
  const float* Wk = (const float*)d_in[3];
  const float* bk = (const float*)d_in[4];
  const float* Wv = (const float*)d_in[5];
  const float* bv = (const float*)d_in[6];
  const float* Wo = (const float*)d_in[7];
  const float* bo = (const float*)d_in[8];
  const float* W1 = (const float*)d_in[9];
  const float* b1 = (const float*)d_in[10];
  const float* W2 = (const float*)d_in[11];
  const float* b2 = (const float*)d_in[12];
  const float* g1 = (const float*)d_in[13];
  const float* g2 = (const float*)d_in[14];
  float* out = (float*)d_out;

  char* ws = (char*)d_ws;
  size_t off = 0;
  auto alloc = [&](size_t bytes) -> void* {
    void* p = ws + off;
    off += (bytes + 255) & ~(size_t)255;
    return p;
  };

  bf16* xb   = (bf16*)alloc((size_t)MTOK * EMB * 2);
  bf16* WqT  = (bf16*)alloc((size_t)EMB * EMB * 2);
  bf16* WkT  = (bf16*)alloc((size_t)KVD * EMB * 2);
  bf16* WvT  = (bf16*)alloc((size_t)KVD * EMB * 2);
  bf16* WoT  = (bf16*)alloc((size_t)EMB * EMB * 2);
  bf16* W1T  = (bf16*)alloc((size_t)FFND * EMB * 2);
  bf16* W2T  = (bf16*)alloc((size_t)EMB * FFND * 2);
  float* qf  = (float*)alloc((size_t)MTOK * EMB * 4);
  float* kf  = (float*)alloc((size_t)MTOK * KVD * 4);
  bf16* qbuf = (bf16*)alloc((size_t)MTOK * EMB * 2);
  bf16* kbuf = (bf16*)alloc((size_t)MTOK * KVD * 2);
  bf16* vbuf = (bf16*)alloc((size_t)MTOK * KVD * 2);
  bf16* ctxb = (bf16*)alloc((size_t)MTOK * EMB * 2);
  float* attn = (float*)alloc((size_t)MTOK * EMB * 4);
  float* x1f  = (float*)alloc((size_t)MTOK * EMB * 4);
  bf16* x1b   = (bf16*)alloc((size_t)MTOK * EMB * 2);
  bf16* hb    = (bf16*)alloc((size_t)MTOK * FFND * 2);
  float* yf   = (float*)alloc((size_t)MTOK * EMB * 4);

  // precision staging
  cvt_f32_bf16<<<2048, 256, 0, stream>>>(x, xb, (long)MTOK * EMB);
  cvt_transpose<<<2048, 256, 0, stream>>>(Wq, WqT, EMB, EMB);
  cvt_transpose<<<2048, 256, 0, stream>>>(Wk, WkT, EMB, KVD);
  cvt_transpose<<<2048, 256, 0, stream>>>(Wv, WvT, EMB, KVD);
  cvt_transpose<<<2048, 256, 0, stream>>>(Wo, WoT, EMB, EMB);
  cvt_transpose<<<4096, 256, 0, stream>>>(W1, W1T, EMB, FFND);
  cvt_transpose<<<4096, 256, 0, stream>>>(W2, W2T, FFND, EMB);

  // QKV projections
  gemm_bf16<<<dim3(EMB / 256, MTOK / 128), 256, 0, stream>>>(
      xb, WqT, bq, nullptr, qf, nullptr, EMB, EMB, 0);
  gemm_bf16<<<dim3(KVD / 256, MTOK / 128), 256, 0, stream>>>(
      xb, WkT, bk, nullptr, kf, nullptr, KVD, EMB, 0);
  gemm_bf16<<<dim3(KVD / 256, MTOK / 128), 256, 0, stream>>>(
      xb, WvT, bv, nullptr, nullptr, vbuf, KVD, EMB, 1);

  rotary_bf16<<<4096, 256, 0, stream>>>(qf, qbuf, QHN);
  rotary_bf16<<<2048, 256, 0, stream>>>(kf, kbuf, KVHN);

  // attention
  flash_attn<<<dim3(TT / 128, BB * QHN), 256, 0, stream>>>(qbuf, kbuf, vbuf, ctxb);

  // output projection + residual
  gemm_bf16<<<dim3(EMB / 256, MTOK / 128), 256, 0, stream>>>(
      ctxb, WoT, bo, x, attn, nullptr, EMB, EMB, 3);
  rmsnorm_k<<<MTOK, 256, 0, stream>>>(attn, g1, x1f, x1b, EMB);

  // FFN
  gemm_bf16<<<dim3(FFND / 256, MTOK / 128), 256, 0, stream>>>(
      x1b, W1T, b1, nullptr, nullptr, hb, FFND, EMB, 2);
  gemm_bf16<<<dim3(EMB / 256, MTOK / 128), 256, 0, stream>>>(
      hb, W2T, b2, x1f, yf, nullptr, EMB, FFND, 3);
  rmsnorm_k<<<MTOK, 256, 0, stream>>>(yf, g2, out, nullptr, EMB);
}
